// Layer_56066503082012
// MI455X (gfx1250) — matrix operand order repaired and checked by reading, not compiled, not run on hardware
//
#include <hip/hip_runtime.h>
#include <hip/hip_bf16.h>

// ---------------------------------------------------------------------------
// Transformer block (LN -> QKV -> causal attn -> proj+res -> LN -> MLP+res)
// for MI455X (gfx1250). All GEMM-shaped math goes through
// v_wmma_f32_16x16x32_bf16 (bf16 in, f32 accumulate). The GEMM K-loop is
// software-pipelined (register double-buffered) so fragment loads for k+32
// overlap the WMMAs for k.
// ---------------------------------------------------------------------------

#define D_MODEL 1024
#define NHEAD   16
#define HDIM    64
#define TCTX    2048
#define BSZ     4
#define MTOT    (BSZ * TCTX)      // 8192 rows
#define DFF     (4 * D_MODEL)     // 4096

typedef unsigned short u16;
typedef __attribute__((ext_vector_type(16))) __bf16 v16bf;
typedef __attribute__((ext_vector_type(8)))  float  v8f;

union Frag {
    v16bf v;
    uint4 q[2];
};

__device__ __forceinline__ u16 f2bf(float f) {
    unsigned int u = __float_as_uint(f);
    unsigned int r = u + 0x7FFFu + ((u >> 16) & 1u);   // round-to-nearest-even
    return (u16)(r >> 16);
}

__device__ __forceinline__ v8f wmma_bf16(v16bf a, v16bf b, v8f c) {
    // 8 args: (neg_a, A, neg_b, B, c_mod, C, reuse_a, reuse_b)
    return __builtin_amdgcn_wmma_f32_16x16x32_bf16(false, a, false, b,
                                                   (short)0, c, false, false);
}

// A-fragment (16x32 bf16, row-major source): lane (M=lane%16, h=lane/16)
// holds K = 16h + j, j = 0..15 contiguous -> two contiguous b128.
__device__ __forceinline__ void lda_frag(Frag& f, const u16* ap) {
    f.q[0] = *(const uint4*)(ap);
    f.q[1] = *(const uint4*)(ap + 8);
}
// B-fragment (32x16 bf16 from BT[N,K] row-major): lane (N=lane%16, h) holds
// K = 16h + j, j = 0..15 contiguous -> two contiguous b128.
__device__ __forceinline__ void ldb_frag(Frag& f, const u16* bp) {
    f.q[0] = *(const uint4*)(bp);
    f.q[1] = *(const uint4*)(bp + 8);
}

// ---------------------------------------------------------------------------
// Core: one wave computes a 16(M) x 64(N) tile of C = A[M,K] * BT[N,K]^T.
// K must be a multiple of 64 (true for all call sites: 1024, 4096).
// Register double-buffered: loads for k+32 are in flight during WMMAs for k.
// ---------------------------------------------------------------------------
__device__ __forceinline__ void gemm_tile(
    const u16* __restrict__ A, int lda,
    const u16* __restrict__ BT, int ldb,
    int K, int mrow, int ncol, int lane,
    const float* __restrict__ bias,
    const float* __restrict__ res, long long ldres,
    float* __restrict__ outF, long long ldF,
    u16* __restrict__ outB, long long bsm, long long bsn,
    int relu)
{
    const int lr = lane & 15;
    const int hf = lane >> 4;

    v8f acc[4];
    v8f zero = {0.f, 0.f, 0.f, 0.f, 0.f, 0.f, 0.f, 0.f};
#pragma unroll
    for (int n = 0; n < 4; ++n) acc[n] = zero;

    const u16* ap = A + (size_t)(mrow + lr) * lda + 16 * hf;
    const u16* bp[4];
#pragma unroll
    for (int n = 0; n < 4; ++n)
        bp[n] = BT + (size_t)(ncol + n * 16 + lr) * ldb + 16 * hf;

    Frag aA, aB, bA[4], bB[4];
    lda_frag(aA, ap);
#pragma unroll
    for (int n = 0; n < 4; ++n) ldb_frag(bA[n], bp[n]);

    for (int k0 = 0; k0 < K; k0 += 64) {
        // issue loads for k0+32 (always in range: k0 <= K-64)
        lda_frag(aB, ap + k0 + 32);
#pragma unroll
        for (int n = 0; n < 4; ++n) ldb_frag(bB[n], bp[n] + k0 + 32);
        if (k0 + 128 < K) __builtin_prefetch(ap + k0 + 128, 0, 1);

        // WMMAs on k0 set (only waits on the older loads)
#pragma unroll
        for (int n = 0; n < 4; ++n) acc[n] = wmma_bf16(aA.v, bA[n].v, acc[n]);

        // issue loads for k0+64 while k0+32 WMMAs run
        if (k0 + 64 < K) {
            lda_frag(aA, ap + k0 + 64);
#pragma unroll
            for (int n = 0; n < 4; ++n) ldb_frag(bA[n], bp[n] + k0 + 64);
        }

#pragma unroll
        for (int n = 0; n < 4; ++n) acc[n] = wmma_bf16(aB.v, bB[n].v, acc[n]);
    }

    // C/D layout: VGPR r, lane L: M = r + 8*(L/16), N = L%16
#pragma unroll
    for (int n = 0; n < 4; ++n) {
        int col = ncol + n * 16 + lr;
        float bv = bias ? bias[col] : 0.f;
#pragma unroll
        for (int r = 0; r < 8; ++r) {
            long long row = mrow + r + 8 * hf;
            float v = acc[n][r] + bv;
            if (res)  v += res[row * ldres + col];
            if (relu) v = v > 0.f ? v : 0.f;
            if (outF) outF[row * ldF + col] = v;
            if (outB) outB[row * bsm + (long long)col * bsn] = f2bf(v);
        }
    }
}

// Generic GEMM: grid = (N/64, M/128), block = 256 (8 waves of 16 rows).
__global__ void gemm_kernel(const u16* __restrict__ A, int lda,
                            const u16* __restrict__ BT, int ldb, int K,
                            const float* __restrict__ bias,
                            const float* __restrict__ res, long long ldres,
                            float* __restrict__ outF, long long ldF,
                            u16* __restrict__ outB, long long bsm, long long bsn,
                            int relu)
{
    int wave = threadIdx.x >> 5;
    int lane = threadIdx.x & 31;
    int mrow = (blockIdx.y * 8 + wave) * 16;
    int ncol = blockIdx.x * 64;
    gemm_tile(A, lda, BT, ldb, K, mrow, ncol, lane,
              bias, res, ldres, outF, ldF, outB, bsm, bsn, relu);
}

// QKV projections. grid = (1, T/128, 3*B*H), block = 256.
// q,k: [B,H,T,HD] bf16 row-major. v: stored TRANSPOSED [B,H,HD,T] bf16.
__global__ void qkv_kernel(const u16* __restrict__ xn,
                           const u16* __restrict__ wT,   // [3][H][HD][D] bf16
                           u16* __restrict__ qb, u16* __restrict__ kb,
                           u16* __restrict__ vtb)
{
    int wave = threadIdx.x >> 5;
    int lane = threadIdx.x & 31;
    int z = blockIdx.z;
    int which = z / (BSZ * NHEAD);
    int bh = z % (BSZ * NHEAD);
    int b = bh / NHEAD;
    int h = bh % NHEAD;

    const u16* Ab  = xn + (size_t)b * TCTX * D_MODEL;
    const u16* BTb = wT + ((size_t)which * NHEAD + h) * HDIM * D_MODEL;
    int mrow = (blockIdx.y * 8 + wave) * 16;  // row t within this batch

    u16* outB;
    long long bsm, bsn;
    if (which == 0)      { outB = qb  + (size_t)bh * TCTX * HDIM; bsm = HDIM; bsn = 1; }
    else if (which == 1) { outB = kb  + (size_t)bh * TCTX * HDIM; bsm = HDIM; bsn = 1; }
    else                 { outB = vtb + (size_t)bh * HDIM * TCTX; bsm = 1;    bsn = TCTX; }

    gemm_tile(Ab, D_MODEL, BTb, D_MODEL, D_MODEL, mrow, 0, lane,
              nullptr, nullptr, 0, nullptr, 0, outB, bsm, bsn, 0);
}

// ---------------------------------------------------------------------------
// Flash attention. grid = (T/128, B*H), block = 256. Each wave: 16 q rows.
// scores = Q K^T / 8 (causal), online softmax, O = P V.
// k is [T,HD] row-major -> B-fragments for QK^T directly. v is [HD,T]
// (transposed) -> B-fragments for PV directly. P re-swizzled via LDS.
// ---------------------------------------------------------------------------
__global__ void attn_kernel(const u16* __restrict__ q,
                            const u16* __restrict__ k,
                            const u16* __restrict__ vT,
                            u16* __restrict__ attn)   // [B*T, D] bf16
{
    __shared__ __align__(16) u16 pls[8][16][32];   // per-wave P tile

    int wave = threadIdx.x >> 5;
    int lane = threadIdx.x & 31;
    int lr = lane & 15;
    int hf = lane >> 4;

    int bh = blockIdx.y;
    int b  = bh / NHEAD;
    int h  = bh % NHEAD;
    int t0 = blockIdx.x * 128 + wave * 16;

    const u16* qb = q  + (size_t)bh * TCTX * HDIM;
    const u16* kb = k  + (size_t)bh * TCTX * HDIM;
    const u16* vb = vT + (size_t)bh * HDIM * TCTX;

    // Q fragments for K-dim chunks 0..31 and 32..63 (loaded once)
    Frag qf[2];
    const u16* qp = qb + (size_t)(t0 + lr) * HDIM + 16 * hf;
    qf[0].q[0] = *(const uint4*)(qp);
    qf[0].q[1] = *(const uint4*)(qp + 8);
    qf[1].q[0] = *(const uint4*)(qp + 32);
    qf[1].q[1] = *(const uint4*)(qp + 40);

    v8f o[4];
    v8f zero = {0.f, 0.f, 0.f, 0.f, 0.f, 0.f, 0.f, 0.f};
#pragma unroll
    for (int e = 0; e < 4; ++e) o[e] = zero;

    float m[8], l[8];
#pragma unroll
    for (int r = 0; r < 8; ++r) { m[r] = -__builtin_inff(); l[r] = 0.f; }

    const float NEGINF = -__builtin_inff();
    int send = t0 + 15;

    for (int s0 = 0; s0 <= send; s0 += 32) {
        // ---- scores: two 16x16 accumulators over 32 s-columns ----
        v8f sc[2];
        sc[0] = zero; sc[1] = zero;
#pragma unroll
        for (int half = 0; half < 2; ++half) {
#pragma unroll
            for (int kk = 0; kk < 2; ++kk) {
                Frag bf_;
                const u16* bp = kb + (size_t)(s0 + half * 16 + lr) * HDIM
                                   + kk * 32 + 16 * hf;
                bf_.q[0] = *(const uint4*)(bp);
                bf_.q[1] = *(const uint4*)(bp + 8);
                sc[half] = wmma_bf16(qf[kk].v, bf_.v, sc[half]);
            }
        }

        // ---- online softmax update (rows r+8h per lane, col = lane%16) ----
        float p0[8], p1[8];
#pragma unroll
        for (int r = 0; r < 8; ++r) {
            int t = t0 + r + 8 * hf;
            float x0 = (s0 + lr      <= t) ? sc[0][r] * 0.125f : NEGINF;
            float x1 = (s0 + 16 + lr <= t) ? sc[1][r] * 0.125f : NEGINF;
            float mx = fmaxf(x0, x1);
#pragma unroll
            for (int d = 1; d < 16; d <<= 1) mx = fmaxf(mx, __shfl_xor(mx, d, 32));
            float mnew = fmaxf(m[r], mx);
            float sfac = __expf(m[r] - mnew);       // 0 on first chunk
            p0[r] = __expf(x0 - mnew);
            p1[r] = __expf(x1 - mnew);
            float rs = p0[r] + p1[r];
#pragma unroll
            for (int d = 1; d < 16; d <<= 1) rs += __shfl_xor(rs, d, 32);
            l[r] = l[r] * sfac + rs;
            m[r] = mnew;
#pragma unroll
            for (int e = 0; e < 4; ++e) o[e][r] *= sfac;
        }

        // ---- P (C-layout) -> LDS -> A-fragment layout ----
#pragma unroll
        for (int r = 0; r < 8; ++r) {
            int rl = r + 8 * hf;
            pls[wave][rl][lr]      = f2bf(p0[r]);
            pls[wave][rl][16 + lr] = f2bf(p1[r]);
        }
        asm volatile("s_wait_dscnt 0" ::: "memory");   // wave-local RAW fence
        Frag pf;
        pf.q[0] = *(const uint4*)&pls[wave][lr][16 * hf];
        pf.q[1] = *(const uint4*)&pls[wave][lr][16 * hf + 8];
        asm volatile("s_wait_dscnt 0" ::: "memory");   // before next overwrite

        // ---- O += P * V  (V^T rows are output cols; K = s chunk) ----
#pragma unroll
        for (int e = 0; e < 4; ++e) {
            Frag bf_;
            const u16* bp = vb + (size_t)(e * 16 + lr) * TCTX + s0 + 16 * hf;
            bf_.q[0] = *(const uint4*)(bp);
            bf_.q[1] = *(const uint4*)(bp + 8);
            o[e] = wmma_bf16(pf.v, bf_.v, o[e]);
        }
    }

    // ---- epilogue: attn[b*T+t, h*64 + e] = O / l ----
    size_t obase = ((size_t)b * TCTX) * D_MODEL + (size_t)h * HDIM;
#pragma unroll
    for (int e = 0; e < 4; ++e) {
#pragma unroll
        for (int r = 0; r < 8; ++r) {
            int t = t0 + r + 8 * hf;
            attn[obase + (size_t)t * D_MODEL + e * 16 + lr] = f2bf(o[e][r] / l[r]);
        }
    }
}

// ---------------------------------------------------------------------------
// Helpers: weight convert+transpose (f32 [R,C] -> bf16 [C,R]) and LayerNorm.
// ---------------------------------------------------------------------------
__global__ void conv_t_kernel(const float* __restrict__ in, u16* __restrict__ out,
                              int R, int C)
{
    size_t slab = (size_t)blockIdx.y * R * C;
    in += slab; out += slab;
    int total = R * C;
    for (int idx = blockIdx.x * blockDim.x + threadIdx.x; idx < total;
         idx += gridDim.x * blockDim.x) {
        int r = idx / C, c = idx % C;
        out[(size_t)c * R + r] = f2bf(in[idx]);
    }
}

__global__ void ln_kernel(const float* __restrict__ x,
                          const float* __restrict__ w,
                          const float* __restrict__ bsh,
                          u16* __restrict__ out)
{
    __shared__ float red[256];
    int row = blockIdx.x;
    int tid = threadIdx.x;
    const float* xr = x + (size_t)row * D_MODEL;

    float vals[4];
    float s = 0.f;
#pragma unroll
    for (int i = 0; i < 4; ++i) { vals[i] = xr[tid + i * 256]; s += vals[i]; }
    red[tid] = s; __syncthreads();
    for (int st = 128; st > 0; st >>= 1) {
        if (tid < st) red[tid] += red[tid + st];
        __syncthreads();
    }
    float mu = red[0] * (1.f / D_MODEL);
    __syncthreads();

    float s2 = 0.f;
#pragma unroll
    for (int i = 0; i < 4; ++i) { float d = vals[i] - mu; s2 += d * d; }
    red[tid] = s2; __syncthreads();
    for (int st = 128; st > 0; st >>= 1) {
        if (tid < st) red[tid] += red[tid + st];
        __syncthreads();
    }
    float inv = rsqrtf(red[0] * (1.f / D_MODEL) + 1e-5f);

#pragma unroll
    for (int i = 0; i < 4; ++i) {
        int c = tid + i * 256;
        out[(size_t)row * D_MODEL + c] = f2bf((vals[i] - mu) * inv * w[c] + bsh[c]);
    }
}

// ---------------------------------------------------------------------------
extern "C" void kernel_launch(void* const* d_in, const int* in_sizes, int n_in,
                              void* d_out, int out_size, void* d_ws, size_t ws_size,
                              hipStream_t stream)
{
    const float* x      = (const float*)d_in[0];
    const float* ln1_w  = (const float*)d_in[1];
    const float* ln1_b  = (const float*)d_in[2];
    const float* wq     = (const float*)d_in[3];
    const float* wk     = (const float*)d_in[4];
    const float* wv     = (const float*)d_in[5];
    const float* w_proj = (const float*)d_in[6];
    const float* b_proj = (const float*)d_in[7];
    const float* ln2_w  = (const float*)d_in[8];
    const float* ln2_b  = (const float*)d_in[9];
    const float* w1     = (const float*)d_in[10];
    const float* b1     = (const float*)d_in[11];
    const float* w2     = (const float*)d_in[12];
    const float* b2     = (const float*)d_in[13];
    float* out = (float*)d_out;

    // ---- workspace layout (~136 MB) ----
    char* p = (char*)d_ws;
    auto alloc = [&](size_t bytes) -> char* {
        char* r = p;
        p += (bytes + 255) & ~(size_t)255;
        return r;
    };
    u16* wqkvT = (u16*)alloc((size_t)3 * NHEAD * HDIM * D_MODEL * 2); // 6 MB
    u16* wprojT = (u16*)alloc((size_t)D_MODEL * D_MODEL * 2);         // 2 MB
    u16* w1T   = (u16*)alloc((size_t)D_MODEL * DFF * 2);              // 8 MB
    u16* w2T   = (u16*)alloc((size_t)DFF * D_MODEL * 2);              // 8 MB
    u16* xn    = (u16*)alloc((size_t)MTOT * D_MODEL * 2);             // 16 MB (reused for xn2)
    char* big  = alloc((size_t)4 * MTOT * D_MODEL * 2);               // 64 MB
    u16* qbuf  = (u16*)big;                                           // q,k,vT,attn ...
    u16* kbuf  = qbuf + (size_t)MTOT * HDIM * NHEAD;                  //  ... reused as
    u16* vtbuf = kbuf + (size_t)MTOT * HDIM * NHEAD;                  //  ... FFN hidden
    u16* attnb = vtbuf + (size_t)MTOT * HDIM * NHEAD;
    u16* hbuf  = (u16*)big;                                           // [8192, 4096] bf16
    float* x1  = (float*)alloc((size_t)MTOT * D_MODEL * 4);           // 32 MB

    dim3 blk(256);

    // 1) weight convert + transpose to bf16 [N,K]
    conv_t_kernel<<<dim3(64, NHEAD), blk, 0, stream>>>(wq, wqkvT, D_MODEL, HDIM);
    conv_t_kernel<<<dim3(64, NHEAD), blk, 0, stream>>>(
        wk, wqkvT + (size_t)NHEAD * HDIM * D_MODEL, D_MODEL, HDIM);
    conv_t_kernel<<<dim3(64, NHEAD), blk, 0, stream>>>(
        wv, wqkvT + (size_t)2 * NHEAD * HDIM * D_MODEL, D_MODEL, HDIM);
    conv_t_kernel<<<dim3(256, 1), blk, 0, stream>>>(w_proj, wprojT, D_MODEL, D_MODEL);
    conv_t_kernel<<<dim3(1024, 1), blk, 0, stream>>>(w1, w1T, D_MODEL, DFF);
    conv_t_kernel<<<dim3(1024, 1), blk, 0, stream>>>(w2, w2T, DFF, D_MODEL);

    // 2) LN1 -> xn (bf16)
    ln_kernel<<<dim3(MTOT), blk, 0, stream>>>(x, ln1_w, ln1_b, xn);

    // 3) QKV projections (WMMA)
    qkv_kernel<<<dim3(1, TCTX / 128, 3 * BSZ * NHEAD), blk, 0, stream>>>(
        xn, wqkvT, qbuf, kbuf, vtbuf);

    // 4) causal flash attention (WMMA)
    attn_kernel<<<dim3(TCTX / 128, BSZ * NHEAD), blk, 0, stream>>>(
        qbuf, kbuf, vtbuf, attnb);

    // 5) output projection + bias + residual -> x1 (f32)
    gemm_kernel<<<dim3(D_MODEL / 64, MTOT / 128), blk, 0, stream>>>(
        attnb, D_MODEL, wprojT, D_MODEL, D_MODEL,
        b_proj, x, D_MODEL, x1, D_MODEL, nullptr, 0, 0, 0);

    // 6) LN2 -> xn (reuse, bf16)
    ln_kernel<<<dim3(MTOT), blk, 0, stream>>>(x1, ln2_w, ln2_b, xn);

    // 7) FFN up + ReLU -> hbuf (bf16)  [overwrites q/k/v/attn region]
    gemm_kernel<<<dim3(DFF / 64, MTOT / 128), blk, 0, stream>>>(
        xn, D_MODEL, w1T, D_MODEL, D_MODEL,
        b1, nullptr, 0, nullptr, 0, hbuf, DFF, 1, 1);

    // 8) FFN down + bias + residual -> d_out (f32)
    gemm_kernel<<<dim3(D_MODEL / 64, MTOT / 128), blk, 0, stream>>>(
        hbuf, DFF, w2T, DFF, DFF,
        b2, x1, D_MODEL, out, D_MODEL, nullptr, 0, 0, 0);
}
